// InteractionNetwork_23613730194127
// MI455X (gfx1250) — matrix operand order repaired and checked by reading, not compiled, not run on hardware
//
#include <hip/hip_runtime.h>

typedef __attribute__((ext_vector_type(16))) _Float16 v16h;
typedef __attribute__((ext_vector_type(8)))  float    v8f;
typedef __attribute__((ext_vector_type(4)))  unsigned u32x4;

union FragH {
    u32x4 q[2];
    v16h  h;
};

__device__ __forceinline__ unsigned pack_h2(float a, float b) {
    _Float16 ha = (_Float16)a, hb = (_Float16)b;
    unsigned short ua = __builtin_bit_cast(unsigned short, ha);
    unsigned short ub = __builtin_bit_cast(unsigned short, hb);
    return (unsigned)ua | ((unsigned)ub << 16);
}

__device__ __forceinline__ void wave_lds_fence() {
    asm volatile("s_wait_dscnt 0" ::: "memory");
}

// ---------------------------------------------------------------------------
// One MLP layer, M-blocked: D((MT*16)xN) = act( A * W + b )
// A layout (ISA 7.12.2, 16-bit A 16x32): lane m=l&15, h=l>>4,
//   regs 0-3: K = 8h+2r,+1 ; regs 4-7: K = 16+8h+2(r-4),+1  (per 32-K tile)
// B frags prepacked: u32 t = tile*256 + lane*8 + r holds {W[k0,n], W[k0+1,n]}
//   with n = nt*16 + (lane&15), k0 = kt*32 + 8*(l>>4) + 2r + (r>=4 ? 8 : 0)
// Each B fragment is loaded once per (nt,kt) and reused for MT row-tiles.
// ---------------------------------------------------------------------------
template<int K, int NT, int MT, bool SILU, bool OUT16>
__device__ __forceinline__ void mlp_layer(
    const unsigned short* __restrict__ xbuf,   // LDS: (MT*16) x K f16, row-major
    const u32x4* __restrict__ wfrag,           // global: packed B fragments
    const float* __restrict__ bias,            // global: N floats
    unsigned short* __restrict__ of16,         // LDS out (f16) if OUT16
    float* __restrict__ of32,                  // LDS out (f32) if !OUT16
    int lane)
{
    const int KT = K / 32;
    const int N  = NT * 16;
    const int m  = lane & 15;
    const int hh = lane >> 4;

    #pragma unroll 1
    for (int nt = 0; nt < NT; ++nt) {
        float bv = bias[nt * 16 + m];
        v8f acc[MT];
        #pragma unroll
        for (int mt = 0; mt < MT; ++mt) {
            v8f z = {bv, bv, bv, bv, bv, bv, bv, bv};
            acc[mt] = z;
        }

        #pragma unroll
        for (int kt = 0; kt < KT; ++kt) {
            FragH b;
            const u32x4* bp = wfrag + (size_t)(kt * NT + nt) * 64 + lane * 2;
            b.q[0] = bp[0];
            b.q[1] = bp[1];
            #pragma unroll
            for (int mt = 0; mt < MT; ++mt) {
                FragH a;
                const u32x4* ap = reinterpret_cast<const u32x4*>(
                    xbuf + (mt * 16 + m) * K + kt * 32 + 8 * hh);
                a.q[0] = ap[0];   // regs 0-3: K = 8h .. 8h+7   (within tile)
                a.q[1] = ap[2];   // regs 4-7: K = 16+8h .. 16+8h+7
                acc[mt] = __builtin_amdgcn_wmma_f32_16x16x32_f16(
                    false, a.h, false, b.h, (short)0, acc[mt], false, false);
            }
        }

        #pragma unroll
        for (int mt = 0; mt < MT; ++mt) {
            #pragma unroll
            for (int r = 0; r < 8; ++r) {
                float x = acc[mt][r];
                if (SILU) x = x / (1.0f + __expf(-x));
                int mo = mt * 16 + r + 8 * hh;  // C/D: row = reg + 8*half
                int n  = nt * 16 + m;           //      col = lane&15
                if (OUT16) {
                    _Float16 hv = (_Float16)x;
                    of16[mo * N + n] = __builtin_bit_cast(unsigned short, hv);
                } else {
                    of32[mo * N + n] = x;
                }
            }
        }
    }
}

// ---------------------------------------------------------------------------
// Weight packer: f32 W[KxN] -> f16-pair B fragments (layout described above)
// ---------------------------------------------------------------------------
__global__ void pack_w_kernel(const float* __restrict__ W,
                              unsigned* __restrict__ dst, int K, int N) {
    int total = (K * N) >> 1;
    int NT = N >> 4;
    for (int t = blockIdx.x * blockDim.x + threadIdx.x; t < total;
         t += gridDim.x * blockDim.x) {
        int r    = t & 7;
        int lane = (t >> 3) & 31;
        int tile = t >> 8;
        int kt = tile / NT;
        int nt = tile - kt * NT;
        int hh = lane >> 4;
        int n  = nt * 16 + (lane & 15);
        int k0 = kt * 32 + hh * 8 + 2 * r + (r >= 4 ? 8 : 0);
        dst[t] = pack_h2(W[(size_t)k0 * N + n], W[(size_t)(k0 + 1) * N + n]);
    }
}

__global__ void zero_kernel(float* __restrict__ p, long long n) {
    for (long long i = blockIdx.x * (long long)blockDim.x + threadIdx.x; i < n;
         i += (long long)gridDim.x * blockDim.x)
        p[i] = 0.0f;
}

// ---------------------------------------------------------------------------
// Edge kernel: one wave per 32-edge job (2 row-tiles, shared B fragments).
//   x = [edges | nodes[snd] | nodes[rcv]]  (32 x 384)
//   y = LN(MLP(x));  out_edges = edges + y;  agg[rcv] += y (f32 atomics)
// ---------------------------------------------------------------------------
#define EW 2
__global__ __launch_bounds__(EW * 32) void in_edge_kernel(
    const float* __restrict__ edges, const float* __restrict__ nodes,
    const int* __restrict__ senders, const int* __restrict__ receivers,
    const u32x4* __restrict__ w0, const float* __restrict__ b0,
    const u32x4* __restrict__ w1, const float* __restrict__ b1,
    const u32x4* __restrict__ w2, const float* __restrict__ b2,
    const float* __restrict__ lng, const float* __restrict__ lnb,
    float* __restrict__ agg, float* __restrict__ out_edges,
    int E, int njobs)
{
    // per-wave: xbuf 32x384 f16 (24KB) + hbuf 32x256 f16 (16KB) = 40KB
    __shared__ __align__(16) unsigned short smem[EW * 20480];
    int wave = threadIdx.x >> 5, lane = threadIdx.x & 31;
    int job = blockIdx.x * EW + wave;
    if (job >= njobs) return;

    unsigned short* xbuf = smem + wave * 20480;
    unsigned short* hbuf = xbuf + 12288;
    float* ybuf = reinterpret_cast<float*>(hbuf);   // 32x128 f32, reuses hbuf
    int e0 = job * 32;

    // ---- stage edge_in into LDS as f16 (u32-pair granularity) ----
    for (int t = lane; t < 32 * 192; t += 32) {
        int row = t / 192, pc = t - row * 192, col = pc * 2;
        int er = e0 + row;
        if (er >= E) er = E - 1;   // clamp (reads only)
        const float* src;
        if (col < 128) {
            src = edges + (size_t)er * 128 + col;
        } else if (col < 256) {
            int s = senders[er];
            src = nodes + (size_t)s * 128 + (col - 128);
        } else {
            int rr = receivers[er];
            src = nodes + (size_t)rr * 128 + (col - 256);
        }
        float2 v = *reinterpret_cast<const float2*>(src);
        reinterpret_cast<unsigned*>(xbuf)[t] = pack_h2(v.x, v.y);
    }
    wave_lds_fence();

    // ---- MLP: 384 -> 256 (silu) -> 256 (silu) -> 128 ----
    mlp_layer<384, 16, 2, true,  true >(xbuf, w0, b0, hbuf, nullptr, lane);
    wave_lds_fence();
    mlp_layer<256, 16, 2, true,  true >(hbuf, w1, b1, xbuf, nullptr, lane);
    wave_lds_fence();
    mlp_layer<256,  8, 2, false, false>(xbuf, w2, b2, nullptr, ybuf, lane);
    wave_lds_fence();

    // ---- LayerNorm + residual + scatter-add (one lane per row) ----
    int e = e0 + lane;
    if (e < E) {
        const float* yr = ybuf + lane * 128;
        float s = 0.0f;
        #pragma unroll 4
        for (int c = 0; c < 128; ++c) s += yr[c];
        float mean = s * (1.0f / 128.0f);
        float var = 0.0f;
        #pragma unroll 4
        for (int c = 0; c < 128; ++c) { float d = yr[c] - mean; var += d * d; }
        float rs = rsqrtf(var * (1.0f / 128.0f) + 1e-5f);

        int rcv = receivers[e];
        const float* erow = edges + (size_t)e * 128;
        float* orow = out_edges + (size_t)e * 128;
        float* arow = agg + (size_t)rcv * 128;
        #pragma unroll 4
        for (int c = 0; c < 128; ++c) {
            float val = (yr[c] - mean) * rs * lng[c] + lnb[c];
            orow[c] = erow[c] + val;
            atomicAdd(arow + c, val);
        }
    }
}

// ---------------------------------------------------------------------------
// Node kernel: one wave per 32-node job.  x = [nodes | agg]  (32 x 256)
// ---------------------------------------------------------------------------
#define NW 2
__global__ __launch_bounds__(NW * 32) void in_node_kernel(
    const float* __restrict__ nodes, const float* __restrict__ agg,
    const u32x4* __restrict__ w0, const float* __restrict__ b0,
    const u32x4* __restrict__ w1, const float* __restrict__ b1,
    const u32x4* __restrict__ w2, const float* __restrict__ b2,
    const float* __restrict__ lng, const float* __restrict__ lnb,
    float* __restrict__ out_nodes, int Nn, int njobs)
{
    // per-wave: xbuf 32x256 f16 (16KB) + hbuf 32x256 f16 (16KB) = 32KB
    __shared__ __align__(16) unsigned short smem[NW * 16384];
    int wave = threadIdx.x >> 5, lane = threadIdx.x & 31;
    int job = blockIdx.x * NW + wave;
    if (job >= njobs) return;

    unsigned short* xbuf = smem + wave * 16384;
    unsigned short* hbuf = xbuf + 8192;
    float* ybuf = reinterpret_cast<float*>(hbuf);   // 32x128 f32, reuses hbuf
    int i0 = job * 32;

    for (int t = lane; t < 32 * 128; t += 32) {
        int row = t >> 7, pc = t & 127, col = pc * 2;
        int ri = i0 + row;
        if (ri >= Nn) ri = Nn - 1;   // clamp (reads only)
        const float* src = (col < 128)
            ? nodes + (size_t)ri * 128 + col
            : agg + (size_t)ri * 128 + (col - 128);
        float2 v = *reinterpret_cast<const float2*>(src);
        reinterpret_cast<unsigned*>(xbuf)[t] = pack_h2(v.x, v.y);
    }
    wave_lds_fence();

    mlp_layer<256, 16, 2, true,  true >(xbuf, w0, b0, hbuf, nullptr, lane);
    wave_lds_fence();
    mlp_layer<256, 16, 2, true,  true >(hbuf, w1, b1, xbuf, nullptr, lane);
    wave_lds_fence();
    mlp_layer<256,  8, 2, false, false>(xbuf, w2, b2, nullptr, ybuf, lane);
    wave_lds_fence();

    int i = i0 + lane;
    if (i < Nn) {
        const float* yr = ybuf + lane * 128;
        float s = 0.0f;
        #pragma unroll 4
        for (int c = 0; c < 128; ++c) s += yr[c];
        float mean = s * (1.0f / 128.0f);
        float var = 0.0f;
        #pragma unroll 4
        for (int c = 0; c < 128; ++c) { float d = yr[c] - mean; var += d * d; }
        float rs = rsqrtf(var * (1.0f / 128.0f) + 1e-5f);

        const float* nrow = nodes + (size_t)i * 128;
        float* orow = out_nodes + (size_t)i * 128;
        #pragma unroll 4
        for (int c = 0; c < 128; ++c) {
            float val = (yr[c] - mean) * rs * lng[c] + lnb[c];
            orow[c] = nrow[c] + val;
        }
    }
}

// ---------------------------------------------------------------------------
extern "C" void kernel_launch(void* const* d_in, const int* in_sizes, int n_in,
                              void* d_out, int out_size, void* d_ws, size_t ws_size,
                              hipStream_t stream) {
    const float* nodes     = (const float*)d_in[0];
    const float* edges     = (const float*)d_in[1];
    const int*   senders   = (const int*)d_in[2];
    const int*   receivers = (const int*)d_in[3];
    const float* ew0 = (const float*)d_in[4];
    const float* eb0 = (const float*)d_in[5];
    const float* ew1 = (const float*)d_in[6];
    const float* eb1 = (const float*)d_in[7];
    const float* ew2 = (const float*)d_in[8];
    const float* eb2 = (const float*)d_in[9];
    const float* elng = (const float*)d_in[10];
    const float* elnb = (const float*)d_in[11];
    const float* nw0 = (const float*)d_in[12];
    const float* nb0 = (const float*)d_in[13];
    const float* nw1 = (const float*)d_in[14];
    const float* nb1 = (const float*)d_in[15];
    const float* nw2 = (const float*)d_in[16];
    const float* nb2 = (const float*)d_in[17];
    const float* nlng = (const float*)d_in[18];
    const float* nlnb = (const float*)d_in[19];

    const int D = 128;
    int N = in_sizes[0] / D;
    int E = in_sizes[1] / D;

    // workspace layout: [agg: N*D f32][packed f16 weight fragments]
    float* agg = (float*)d_ws;
    unsigned* few0 = (unsigned*)((char*)d_ws + (size_t)N * D * sizeof(float));
    unsigned* few1 = few0 + (384 * 256 / 2);
    unsigned* few2 = few1 + (256 * 256 / 2);
    unsigned* fnw0 = few2 + (256 * 128 / 2);
    unsigned* fnw1 = fnw0 + (256 * 256 / 2);
    unsigned* fnw2 = fnw1 + (256 * 256 / 2);

    zero_kernel<<<256, 256, 0, stream>>>(agg, (long long)N * D);
    pack_w_kernel<<<96, 256, 0, stream>>>(ew0, few0, 384, 256);
    pack_w_kernel<<<64, 256, 0, stream>>>(ew1, few1, 256, 256);
    pack_w_kernel<<<32, 256, 0, stream>>>(ew2, few2, 256, 128);
    pack_w_kernel<<<64, 256, 0, stream>>>(nw0, fnw0, 256, 256);
    pack_w_kernel<<<64, 256, 0, stream>>>(nw1, fnw1, 256, 256);
    pack_w_kernel<<<32, 256, 0, stream>>>(nw2, fnw2, 256, 128);

    float* out_nodes = (float*)d_out;
    float* out_edges = (float*)d_out + (size_t)N * D;

    int ejobs = (E + 31) / 32;   // E = 500000 -> 15625 exact
    in_edge_kernel<<<(ejobs + EW - 1) / EW, EW * 32, 0, stream>>>(
        edges, nodes, senders, receivers,
        (const u32x4*)few0, eb0, (const u32x4*)few1, eb1, (const u32x4*)few2, eb2,
        elng, elnb, agg, out_edges, E, ejobs);

    int njobs = (N + 31) / 32;   // N = 50000 -> 1563 (last job: 16 rows)
    in_node_kernel<<<(njobs + NW - 1) / NW, NW * 32, 0, stream>>>(
        nodes, agg,
        (const u32x4*)fnw0, nb0, (const u32x4*)fnw1, nb1, (const u32x4*)fnw2, nb2,
        nlng, nlnb, out_nodes, N, njobs);
}
